// LSTM_7739531067941
// MI455X (gfx1250) — matrix operand order repaired and checked by reading, not compiled, not run on hardware
//
#include <hip/hip_runtime.h>
#include <hip/hip_bf16.h>

#define DEVINL __device__ __forceinline__

#if defined(__has_builtin)
#if __has_builtin(__builtin_amdgcn_sched_barrier)
#define SCHED_BARRIER() __builtin_amdgcn_sched_barrier(0)
#endif
#endif
#ifndef SCHED_BARRIER
#define SCHED_BARRIER() asm volatile("" ::: "memory")
#endif

typedef __attribute__((ext_vector_type(16))) __bf16 v16bf;
typedef __attribute__((ext_vector_type(8)))  float  v8f;

union Frag16 { v16bf v; uint4 q[2]; };

static constexpr int Bsz = 128, Ssz = 2048, Isz = 256, Hsz = 512, Osz = 256;
static constexpr int Kcat = Isz + Hsz;        // 768 : unified K (x | h)
static constexpr int KSTEPS = Kcat / 32;      // 24
static constexpr int LDS_STRIDE = Kcat + 8;   // 776 elems -> 388 dwords, 388 % 64 = 4 (no bank conflicts)

// ---------- helpers ----------
DEVINL unsigned short f2bf(float f) {
  unsigned u = __float_as_uint(f);
  unsigned r = u + 0x7FFFu + ((u >> 16) & 1u);   // round-to-nearest-even
  return (unsigned short)(r >> 16);
}
DEVINL float sigmoid_f(float x) { return 1.0f / (1.0f + __expf(-x)); }
DEVINL float tanh_f(float x) {
  float t = __expf(-2.0f * fabsf(x));
  float r = (1.0f - t) / (1.0f + t);
  return copysignf(r, x);
}

// B-fragment (32x16 bf16): column N = row n of row-major W; lane l holds
// N = l&15, K = k0 + (l>>4)*16 + j  -> 32 contiguous bytes of W row n.
DEVINL v16bf load_b_frag(const __bf16* __restrict__ Wrow, int k0, int lg) {
  Frag16 f;
  const __bf16* p = Wrow + k0 + lg * 16;
  f.q[0] = *reinterpret_cast<const uint4*>(p);
  f.q[1] = *reinterpret_cast<const uint4*>(p + 8);
  return f.v;
}
// A-fragment from a row-major panel (global or LDS): lane l -> row M = l&15,
// half lg = l>>4; elems j: K=k0+16lg+j.
DEVINL v16bf load_a_frag(const __bf16* __restrict__ row, int k0, int lg) {
  Frag16 f;
  f.q[0] = *reinterpret_cast<const uint4*>(row + k0 + lg * 16);
  f.q[1] = *reinterpret_cast<const uint4*>(row + k0 + lg * 16 + 8);
  return f.v;
}
DEVINL v8f wmma_bf16(v16bf a, v16bf b, v8f c) {
  return __builtin_amdgcn_wmma_f32_16x16x32_bf16(false, a, false, b, (short)0, c,
                                                 false, false);
}

// ---------- fp32 -> bf16 conversion, contiguous (4 elems / thread) ----------
__global__ __launch_bounds__(256) void cvt_f32_bf16(const float* __restrict__ src,
                                                    __bf16* __restrict__ dst, int n4) {
  int i = blockIdx.x * 256 + threadIdx.x;
  if (i >= n4) return;
  float4 v = reinterpret_cast<const float4*>(src)[i];
  ushort4 o;
  o.x = f2bf(v.x); o.y = f2bf(v.y); o.z = f2bf(v.z); o.w = f2bf(v.w);
  reinterpret_cast<ushort4*>(dst)[i] = o;
}

// ---------- fp32 -> bf16 conversion into strided destination rows ----------
__global__ __launch_bounds__(256) void cvt_f32_bf16_2d(const float* __restrict__ src,
                                                       __bf16* __restrict__ dst,
                                                       int cols4, int dstride, int n4) {
  int i = blockIdx.x * 256 + threadIdx.x;
  if (i >= n4) return;
  int r = i / cols4, c = i - r * cols4;
  float4 v = reinterpret_cast<const float4*>(src)[i];
  ushort4 o;
  o.x = f2bf(v.x); o.y = f2bf(v.y); o.z = f2bf(v.z); o.w = f2bf(v.w);
  reinterpret_cast<ushort4*>(dst + (size_t)r * dstride)[c] = o;
}

// ---------- zero h0 (bf16) and c0 (f32) ----------
__global__ __launch_bounds__(256) void init_state(__bf16* __restrict__ h0,
                                                  float* __restrict__ c, int n) {
  int i = blockIdx.x * 256 + threadIdx.x;
  if (i < n) {
    reinterpret_cast<unsigned short*>(h0)[i] = 0;
    c[i] = 0.0f;
  }
}

// ---------- one LSTM timestep ----------
// 32 blocks x 256 thr (8 waves). Block b: m_tile = b>>2 (16 A-rows staged in
// LDS, shared by all 8 waves); wave w: n_tile = (b&3)*8 + w. Each wave owns a
// 16x16 (m,n) tile, computes all 4 gate pre-activations with a software-
// pipelined (depth-2 ping-pong) K loop, then the c/h pointwise update.
__global__ __launch_bounds__(256) void lstm_step(
    const __bf16* __restrict__ x_bf, int t,
    const __bf16* __restrict__ Wcat,   // [4][H][Kcat] bf16 : [Wx | Wh] per gate
    const float* __restrict__ b_i, const float* __restrict__ b_f,
    const float* __restrict__ b_o, const float* __restrict__ b_g,
    const __bf16* __restrict__ h_in, __bf16* __restrict__ h_out,
    float* __restrict__ c_state) {
  __shared__ alignas(16) __bf16 Apanel[16 * LDS_STRIDE];   // ~24.3 KB

  const int m_tile = blockIdx.x >> 2;
  const int wid    = threadIdx.x >> 5;
  const int n_tile = ((blockIdx.x & 3) << 3) | wid;
  const int lane   = threadIdx.x & 31;
  const int m0 = m_tile * 16, n0 = n_tile * 16;
  const int lm = lane & 15, lg = lane >> 4;

  // --- cooperative LDS fill: A panel = [ x_t (16x256) | h (16x512) ] ---
  for (int idx = threadIdx.x; idx < 16 * (Isz / 8); idx += 256) {
    int r = idx >> 5, c = idx & 31;
    uint4 v = *(reinterpret_cast<const uint4*>(
        x_bf + (size_t)(m0 + r) * ((size_t)Ssz * Isz) + (size_t)t * Isz) + c);
    *reinterpret_cast<uint4*>(&Apanel[r * LDS_STRIDE + c * 8]) = v;
  }
  for (int idx = threadIdx.x; idx < 16 * (Hsz / 8); idx += 256) {
    int r = idx >> 6, c = idx & 63;
    uint4 v = *(reinterpret_cast<const uint4*>(h_in + (size_t)(m0 + r) * Hsz) + c);
    *reinterpret_cast<uint4*>(&Apanel[r * LDS_STRIDE + Isz + c * 8]) = v;
  }
  __syncthreads();

  const size_t gstride = (size_t)Hsz * Kcat;
  const __bf16* wrow = Wcat + (size_t)(n0 + lm) * Kcat;
  const __bf16* arow = &Apanel[lm * LDS_STRIDE];

  v8f acc[4] = {};

  // explicit depth-2 ping-pong pipeline; fully unrolled so stage indices fold
  v16bf bufA[2];
  v16bf bufB[2][4];
  bufA[0] = load_a_frag(arow, 0, lg);
#pragma unroll
  for (int g = 0; g < 4; ++g) bufB[0][g] = load_b_frag(wrow + g * gstride, 0, lg);

#pragma unroll
  for (int i = 0; i < KSTEPS; ++i) {
    const int cur = i & 1, nxt = cur ^ 1;
    if (i + 1 < KSTEPS) {
      const int kn = (i + 1) * 32;
      bufA[nxt] = load_a_frag(arow, kn, lg);
#pragma unroll
      for (int g = 0; g < 4; ++g)
        bufB[nxt][g] = load_b_frag(wrow + g * gstride, kn, lg);
    }
    SCHED_BARRIER();   // keep prefetch loads above the WMMAs of step i
#pragma unroll
    for (int g = 0; g < 4; ++g) acc[g] = wmma_bf16(bufA[cur], bufB[cur][g], acc[g]);
    SCHED_BARRIER();
  }

  // --- pointwise LSTM cell update on the accumulator layout ---
  // C/D layout: lane 0-15 -> N=lane, M=r ; lane 16-31 -> N=lane-16, M=r+8.
  const int n = n0 + lm;
  const float bi = b_i[n], bf = b_f[n], bo = b_o[n], bg = b_g[n];
#pragma unroll
  for (int r = 0; r < 8; ++r) {
    const int m = m0 + r + 8 * lg;
    const size_t idx = (size_t)m * Hsz + n;
    float gi = sigmoid_f(acc[0][r] + bi);
    float gf = sigmoid_f(acc[1][r] + bf);
    float go = sigmoid_f(acc[2][r] + bo);
    float gg = tanh_f(acc[3][r] + bg);
    float cn = gg * gi + c_state[idx] * gf;
    c_state[idx] = cn;
    reinterpret_cast<unsigned short*>(h_out)[idx] = f2bf(tanh_f(cn) * go);
  }
}

// ---------- final projection: out = h_final @ W_ph^T + b_ph ----------
__global__ __launch_bounds__(256) void proj_kernel(
    const __bf16* __restrict__ h_bf, const __bf16* __restrict__ Wp,
    const float* __restrict__ b_p, float* __restrict__ out) {
  const int tid    = blockIdx.x * 256 + threadIdx.x;
  const int wave   = tid >> 5;       // 0..127
  const int lane   = threadIdx.x & 31;
  const int m_tile = wave >> 4;      // 0..7
  const int n_tile = wave & 15;      // 0..15
  const int m0 = m_tile * 16, n0 = n_tile * 16;
  const int lm = lane & 15, lg = lane >> 4;

  v8f acc = {};
  const __bf16* arow = h_bf + (size_t)(m0 + lm) * Hsz;
  const __bf16* wrow = Wp + (size_t)(n0 + lm) * Hsz;
#pragma unroll 4
  for (int k0 = 0; k0 < Hsz; k0 += 32) {
    v16bf a = load_a_frag(arow, k0, lg);
    v16bf b = load_b_frag(wrow, k0, lg);
    acc = wmma_bf16(a, b, acc);
  }
  const int n = n0 + lm;
  const float bp = b_p[n];
#pragma unroll
  for (int r = 0; r < 8; ++r) {
    const int m = m0 + r + 8 * lg;
    out[(size_t)m * Osz + n] = acc[r] + bp;
  }
}

// ---------- host ----------
extern "C" void kernel_launch(void* const* d_in, const int* in_sizes, int n_in,
                              void* d_out, int out_size, void* d_ws, size_t ws_size,
                              hipStream_t stream) {
  const float* x    = (const float*)d_in[0];
  const float* W_ix = (const float*)d_in[1];
  const float* W_fx = (const float*)d_in[2];
  const float* W_ox = (const float*)d_in[3];
  const float* W_gx = (const float*)d_in[4];
  const float* W_ih = (const float*)d_in[5];
  const float* b_ih = (const float*)d_in[6];
  const float* W_fh = (const float*)d_in[7];
  const float* b_fh = (const float*)d_in[8];
  const float* W_oh = (const float*)d_in[9];
  const float* b_oh = (const float*)d_in[10];
  const float* W_gh = (const float*)d_in[11];
  const float* b_gh = (const float*)d_in[12];
  const float* W_ph = (const float*)d_in[13];
  const float* b_ph = (const float*)d_in[14];

  char* ws = (char*)d_ws;
  size_t off = 0;
  auto alloc = [&](size_t bytes) -> void* {
    void* p = (void*)(ws + off);
    off = (off + bytes + 255) & ~(size_t)255;
    return p;
  };

  const size_t nX   = (size_t)Bsz * Ssz * Isz;   // 67,108,864
  const size_t gstr = (size_t)Hsz * Kcat;        // per-gate stride in Wcat
  const int    nWp  = Osz * Hsz;                 // 131,072
  const int    nHC  = Bsz * Hsz;                 // 65,536

  __bf16* x_bf  = (__bf16*)alloc(nX * 2);                    // ~134 MB
  __bf16* Wcat  = (__bf16*)alloc((size_t)4 * gstr * 2);      // 3 MB  [4][H][768]
  __bf16* Wp_bf = (__bf16*)alloc((size_t)nWp * 2);
  __bf16* hbuf0 = (__bf16*)alloc((size_t)nHC * 2);
  __bf16* hbuf1 = (__bf16*)alloc((size_t)nHC * 2);
  float*  cbuf  = (float*)alloc((size_t)nHC * 4);

  // x -> bf16 (one-shot)
  cvt_f32_bf16<<<(int)(nX / 4 / 256), 256, 0, stream>>>(x, x_bf, (int)(nX / 4));

  // weights -> concatenated bf16 [Wx | Wh] per gate (L2-resident afterwards)
  const int n4x = Hsz * Isz / 4;   // 32768
  const int n4h = Hsz * Hsz / 4;   // 65536
  const float* Wx_src[4] = {W_ix, W_fx, W_ox, W_gx};
  const float* Wh_src[4] = {W_ih, W_fh, W_oh, W_gh};
  for (int g = 0; g < 4; ++g) {
    cvt_f32_bf16_2d<<<n4x / 256, 256, 0, stream>>>(Wx_src[g], Wcat + g * gstr,
                                                   Isz / 4, Kcat, n4x);
    cvt_f32_bf16_2d<<<n4h / 256, 256, 0, stream>>>(Wh_src[g], Wcat + g * gstr + Isz,
                                                   Hsz / 4, Kcat, n4h);
  }
  cvt_f32_bf16<<<nWp / 4 / 256, 256, 0, stream>>>(W_ph, Wp_bf, nWp / 4);

  init_state<<<nHC / 256, 256, 0, stream>>>(hbuf0, cbuf, nHC);

  // sequential recurrence; h double-buffered, c updated in place (tile-exclusive)
  for (int t = 0; t < Ssz; ++t) {
    const __bf16* hin  = (t & 1) ? hbuf1 : hbuf0;
    __bf16*       hout = (t & 1) ? hbuf0 : hbuf1;
    lstm_step<<<32, 256, 0, stream>>>(x_bf, t, Wcat,
                                      b_ih, b_fh, b_oh, b_gh, hin, hout, cbuf);
  }
  // t = 2047 (odd) wrote hbuf0 -> final hidden state
  proj_kernel<<<16, 256, 0, stream>>>(hbuf0, Wp_bf, b_ph, (float*)d_out);
}
